// Quantize_83219286327492
// MI455X (gfx1250) — matrix operand order repaired and checked by reading, not compiled, not run on hardware
//
#include <hip/hip_runtime.h>
#include <hip/hip_bf16.h>

// ---------------------------------------------------------------------------
// Vector-quantize (nearest + furthest code) for MI455X / gfx1250.
//   x:     [32,4096,64] f32   (N = 131072 rows, dim = 64)
//   embed: [64,2048]    f32
// out = concat( mix_quantize [131072*64] f32 , diff scalar f32 )
// Strategy: bf16 WMMA 16x16x32 for the distance GEMM, full codebook staged in
// LDS (256KB of the 320KB/WGP), fused argmin/argmax + gather + diff.
// ---------------------------------------------------------------------------

typedef __attribute__((ext_vector_type(16))) __bf16 v16bf;
typedef __attribute__((ext_vector_type(8)))  float  v8f;

#define DIM        64
#define NE         2048
#define NROWS      (32 * 4096)          // 131072
#define NCH        (NE / 16)            // 128 column chunks of 16 codes
#define OUT_ELEMS  (NROWS * DIM)        // 8388608
#define SMEM_B_BYTES  (NE * DIM * 2)    // 262144 bytes of packed bf16 B
#define SMEM_BYTES    (SMEM_B_BYTES + NE * 4)  // + c2[2048] = 270336 bytes

// ---- Stage 1: normalize codebook columns -----------------------------------
// codes[j][k] = embed[k][j] / max(||embed[:,j]||, 1e-12) ; c2[j] = ||codes_j||^2
__global__ void vq_normalize_kernel(const float* __restrict__ embed,
                                    float* __restrict__ codes,
                                    float* __restrict__ c2) {
    int j = blockIdx.x * blockDim.x + threadIdx.x;   // 0..2047, exact grid
    float s = 0.0f;
    #pragma unroll
    for (int k = 0; k < DIM; ++k) {
        float v = embed[k * NE + j];                 // coalesced across j
        s += v * v;
    }
    float inv = 1.0f / fmaxf(sqrtf(s), 1e-12f);
    #pragma unroll
    for (int k = 0; k < DIM; ++k)
        codes[j * DIM + k] = embed[k * NE + j] * inv;
    c2[j] = s * inv * inv;
}

// ---- Stage 2: pack B = -2 * codes^T into WMMA bf16 lane layout -------------
// Layout per 16-code chunk c, K-block b (32 K each), lane L (0..31), elem e:
//   j = c*16 + (L&15);  k = 32*b + 16*(L>>4) + e
// so each lane of stage 3 reads one contiguous 32B v16bf per B fragment.
__global__ void vq_pack_kernel(const float* __restrict__ codes,
                               __bf16* __restrict__ bws,
                               float* __restrict__ diff_scalar) {
    int tid = blockIdx.x * blockDim.x + threadIdx.x; // 0..131071, exact grid
    int e = tid & 15;
    int L = (tid >> 4) & 31;
    int b = (tid >> 9) & 1;
    int c = tid >> 10;
    int j = c * 16 + (L & 15);
    int k = 32 * b + 16 * (L >> 4) + e;
    bws[tid] = (__bf16)(-2.0f * codes[j * DIM + k]);
    if (tid == 0) *diff_scalar = 0.0f;               // harness poisons d_out
}

// ---- Stage 3: fused distance-GEMM + argmin/argmax + gather -----------------
__global__ __launch_bounds__(256)
void vq_main_kernel(const float* __restrict__ x,
                    const float* __restrict__ codes,
                    const float* __restrict__ c2,
                    const __bf16* __restrict__ bws,
                    float* __restrict__ out) {
    extern __shared__ char smem[];
    __bf16* sB  = (__bf16*)smem;                       // 256 KB packed B
    float*  sC2 = (float*)(smem + SMEM_B_BYTES);       // 8 KB c2

    const int t = threadIdx.x;

    // Stage entire packed codebook into LDS (256 KB = 16384 uint4).
    {
        const uint4* gB = (const uint4*)bws;
        uint4*       lB = (uint4*)sB;
        for (int i = 0; i < 64; ++i)
            lB[t + i * 256] = gB[t + i * 256];
        #pragma unroll
        for (int i = 0; i < 8; ++i)
            sC2[t + i * 256] = c2[t + i * 256];
    }
    __syncthreads();

    const int lane = t & 31;
    const int wave = t >> 5;
    const int m    = lane & 15;     // A: row-in-tile; B/C/D: column n
    const int h    = lane >> 4;     // lane half

    // Each wave owns two 16-row tiles to amortize LDS B reads over 4 WMMAs.
    const int tile0 = (blockIdx.x * 8 + wave) * 2;

    // ---- Load A fragments (x rows -> bf16, contiguous k = 16h+e order, same as packed B) ----
    // elem e : K = 16h+e ; second block +32.
    v16bf alo[2], ahi[2];
    #pragma unroll
    for (int tt = 0; tt < 2; ++tt) {
        const float* xr = x + (size_t)(tile0 + tt) * 16 * DIM + (size_t)m * DIM;
        #pragma unroll
        for (int e = 0; e < 8; ++e) {
            alo[tt][e]     = (__bf16)xr[16 * h + e];
            alo[tt][e + 8] = (__bf16)xr[16 * h + 8 + e];
            ahi[tt][e]     = (__bf16)xr[32 + 16 * h + e];
            ahi[tt][e + 8] = (__bf16)xr[40 + 16 * h + e];
        }
    }

    float minV[2][8], maxV[2][8];
    int   minJ[2][8], maxJ[2][8];
    #pragma unroll
    for (int tt = 0; tt < 2; ++tt)
        #pragma unroll
        for (int v = 0; v < 8; ++v) {
            minV[tt][v] =  3.402823466e38f;  minJ[tt][v] = 0;
            maxV[tt][v] = -3.402823466e38f;  maxJ[tt][v] = 0;
        }

    const v16bf* sBv = (const v16bf*)sB;   // 64 fragments per chunk (2 blocks x 32 lanes)

    // ---- Main loop over 128 chunks of 16 codes --------------------------
    for (int c = 0; c < NCH; ++c) {
        const float cc = sC2[c * 16 + m];          // ||E_n||^2 for this column
        v16bf b0 = sBv[c * 64 + lane];             // K block 0..31  (LDS)
        v16bf b1 = sBv[c * 64 + 32 + lane];        // K block 32..63 (LDS)
        const int j = c * 16 + m;

        #pragma unroll
        for (int tt = 0; tt < 2; ++tt) {
            v8f acc;
            #pragma unroll
            for (int v = 0; v < 8; ++v) acc[v] = cc;
            // dist-ish = c2[j] - 2 * <x_row, e_j>   (B pre-scaled by -2)
            acc = __builtin_amdgcn_wmma_f32_16x16x32_bf16(
                      false, alo[tt], false, b0, (short)0, acc, false, false);
            acc = __builtin_amdgcn_wmma_f32_16x16x32_bf16(
                      false, ahi[tt], false, b1, (short)0, acc, false, false);
            #pragma unroll
            for (int v = 0; v < 8; ++v) {
                float d = acc[v];
                if (d < minV[tt][v]) { minV[tt][v] = d; minJ[tt][v] = j; }
                if (d > maxV[tt][v]) { maxV[tt][v] = d; maxJ[tt][v] = j; }
            }
        }
    }

    // ---- Cross-lane argmin/argmax within each 16-lane half ---------------
    // (C/D row v lives in lanes 0-15 (M=v) and 16-31 (M=v+8); columns are lanes)
    #pragma unroll
    for (int off = 1; off <= 8; off <<= 1) {
        #pragma unroll
        for (int tt = 0; tt < 2; ++tt)
            #pragma unroll
            for (int v = 0; v < 8; ++v) {
                float ov = __shfl_xor(minV[tt][v], off, 32);
                int   oj = __shfl_xor(minJ[tt][v], off, 32);
                if (ov < minV[tt][v] ||
                    (ov == minV[tt][v] && oj < minJ[tt][v])) {
                    minV[tt][v] = ov; minJ[tt][v] = oj;
                }
                float ow = __shfl_xor(maxV[tt][v], off, 32);
                int   ok = __shfl_xor(maxJ[tt][v], off, 32);
                if (ow > maxV[tt][v] ||
                    (ow == maxV[tt][v] && ok < maxJ[tt][v])) {
                    maxV[tt][v] = ow; maxJ[tt][v] = ok;
                }
            }
    }

    // ---- Gather codes, write mix output, accumulate diff ----------------
    float s = 0.0f;
    #pragma unroll
    for (int tt = 0; tt < 2; ++tt) {
        #pragma unroll
        for (int r = 0; r < 16; ++r) {
            const int hh = r >> 3, v = r & 7;
            const int jm = __shfl(minJ[tt][v], hh * 16, 32);  // nearest code
            const int jx = __shfl(maxJ[tt][v], hh * 16, 32);  // furthest code
            const size_t row = (size_t)(tile0 + tt) * 16 + r;
            const size_t base = row * DIM + (size_t)lane * 2;
            const float2 xv = *(const float2*)(x + base);
            const float2 qm = *(const float2*)(codes + (size_t)jm * DIM + lane * 2);
            const float2 qx = *(const float2*)(codes + (size_t)jx * DIM + lane * 2);
            float2 o;
            o.x = 0.5f * (qm.x + qx.x);
            o.y = 0.5f * (qm.y + qx.y);
            *(float2*)(out + base) = o;
            float d0 = qm.x - xv.x, d1 = qm.y - xv.y;
            float d2 = qx.x - xv.x, d3 = qx.y - xv.y;
            s += d0 * d0 + d1 * d1 + d2 * d2 + d3 * d3;
        }
    }
    #pragma unroll
    for (int off = 1; off < 32; off <<= 1) s += __shfl_xor(s, off, 32);
    if (lane == 0)
        atomicAdd(out + OUT_ELEMS, s * (1.0f / (float)OUT_ELEMS));
}

// ---------------------------------------------------------------------------
extern "C" void kernel_launch(void* const* d_in, const int* in_sizes, int n_in,
                              void* d_out, int out_size, void* d_ws, size_t ws_size,
                              hipStream_t stream) {
    const float* x     = (const float*)d_in[0];   // [32,4096,64]
    const float* embed = (const float*)d_in[1];   // [64,2048]
    float*       out   = (float*)d_out;           // [8388608 + 1]

    // Workspace layout (776 KB total):
    char*   ws    = (char*)d_ws;
    float*  codes = (float*)ws;                                  // 512 KB
    float*  c2    = (float*)(ws + (size_t)NE * DIM * 4);         // 8 KB
    __bf16* bws   = (__bf16*)(ws + (size_t)NE * DIM * 4 + NE * 4); // 256 KB

    vq_normalize_kernel<<<NE / 256, 256, 0, stream>>>(embed, codes, c2);
    vq_pack_kernel<<<(NE * DIM) / 256, 256, 0, stream>>>(codes, bws, out + OUT_ELEMS);

    // 270 KB dynamic LDS (> default 64 KB): opt in every call (deterministic,
    // host-side only, safe under graph capture).
    hipFuncSetAttribute((const void*)vq_main_kernel,
                        hipFuncAttributeMaxDynamicSharedMemorySize, SMEM_BYTES);

    // 8192 row-tiles of 16; each wave handles 2 tiles; 8 waves/block -> 512 blocks.
    vq_main_kernel<<<NROWS / (16 * 2 * 8), 256, SMEM_BYTES, stream>>>(
        x, codes, c2, bws, out);
}
